// GraphAttentionLayer_74294344286768
// MI455X (gfx1250) — matrix operand order repaired and checked by reading, not compiled, not run on hardware
//
#include <hip/hip_runtime.h>
#include <hip/hip_bf16.h>

typedef __attribute__((ext_vector_type(16))) _Float16 v16h;
typedef __attribute__((ext_vector_type(8)))  _Float16 v8h;
typedef __attribute__((ext_vector_type(8)))  float    v8f;

#define B_DIM 16
#define T_DIM 2048
#define F_DIM 128
#define NN    (B_DIM * T_DIM)   // 32768 nodes
#define ALPHA 0.2f
#define MASKV -1000000000.0f

// ---------------------------------------------------------------------------
// Kernel 0: transpose W (128x128 f32) -> Wt so GEMM B-operand reads are
// contiguous runs of K for a fixed output feature.
// ---------------------------------------------------------------------------
__global__ void k_twt(const float* __restrict__ W, float* __restrict__ Wt) {
    int k = blockIdx.x;      // 0..127
    int o = threadIdx.x;     // 0..127
    Wt[(size_t)o * F_DIM + k] = W[(size_t)k * F_DIM + o];
}

// ---------------------------------------------------------------------------
// Kernel 1: h = x @ W via v_wmma_f32_16x16x32_f16.
// One wave computes a 16(row) x 128(feature) tile of h and stores it
// TRANSPOSED as f16: hT[f][node]. Stores are one b128 per (chunk,lane).
// ---------------------------------------------------------------------------
__global__ __launch_bounds__(128) void k_gemm_h(const float* __restrict__ x,
                                                const float* __restrict__ Wt,
                                                _Float16* __restrict__ hT) {
    int wave = (blockIdx.x * blockDim.x + threadIdx.x) >> 5;
    int lane = threadIdx.x & 31;
    int n = lane & 15;        // N index / M index (lane role)
    int g = lane >> 4;        // half-wave group

    int row0  = wave * 16;            // first node of tile
    int myrow = row0 + n;             // A-matrix row for this lane

    v8f acc[8];
#pragma unroll
    for (int c = 0; c < 8; ++c) acc[c] = v8f{};

#pragma unroll
    for (int ks = 0; ks < 4; ++ks) {
        int kb = ks * 32;
        // A operand: x[myrow][kb + K], K = 16g + e
        const float4* xr = (const float4*)(x + (size_t)myrow * F_DIM + kb);
        float4 r0 = xr[4 * g + 0];
        float4 r1 = xr[4 * g + 1];
        float4 r2 = xr[4 * g + 2];
        float4 r3 = xr[4 * g + 3];
        float xa[16] = { r0.x, r0.y, r0.z, r0.w, r1.x, r1.y, r1.z, r1.w,
                         r2.x, r2.y, r2.z, r2.w, r3.x, r3.y, r3.z, r3.w };
        v16h A;
#pragma unroll
        for (int e = 0; e < 16; ++e) A[e] = (_Float16)xa[e];

#pragma unroll
        for (int c = 0; c < 8; ++c) {
            // B operand: Wt[(c*16+n)][kb + 16g + e], e = 2v+h = K-16g
            const float4* wr =
                (const float4*)(Wt + (size_t)(c * 16 + n) * F_DIM + kb + 16 * g);
            float4 w0 = wr[0], w1 = wr[1], w2 = wr[2], w3 = wr[3];
            float wv[16] = { w0.x, w0.y, w0.z, w0.w, w1.x, w1.y, w1.z, w1.w,
                             w2.x, w2.y, w2.z, w2.w, w3.x, w3.y, w3.z, w3.w };
            v16h Bv;
#pragma unroll
            for (int e = 0; e < 16; ++e) Bv[e] = (_Float16)wv[e];
            acc[c] = __builtin_amdgcn_wmma_f32_16x16x32_f16(
                false, A, false, Bv, (short)0, acc[c], false, false);
        }
    }

    // Store transposed: hT[f][node]; lane holds f = c*16+n, rows v+8g.
#pragma unroll
    for (int c = 0; c < 8; ++c) {
        v8h ov;
#pragma unroll
        for (int v = 0; v < 8; ++v) ov[v] = (_Float16)acc[c][v];
        *(v8h*)(hT + (size_t)(c * 16 + n) * NN + row0 + 8 * g) = ov;
    }
}

// ---------------------------------------------------------------------------
// Kernel 2: s1[n] = sum_f h[n][f]*a1[f] ; s2[n] = sum_f h[n][f]*a2[f]
// hT reads are fully coalesced across the block.
// ---------------------------------------------------------------------------
__global__ void k_score(const _Float16* __restrict__ hT,
                        const float* __restrict__ a,
                        float* __restrict__ s1, float* __restrict__ s2) {
    int nid = blockIdx.x * blockDim.x + threadIdx.x;
    float acc1 = 0.f, acc2 = 0.f;
#pragma unroll 4
    for (int f = 0; f < F_DIM; ++f) {
        float hv = (float)hT[(size_t)f * NN + nid];
        acc1 += hv * a[f];
        acc2 += hv * a[F_DIM + f];
    }
    s1[nid] = acc1;
    s2[nid] = acc2;
}

// ---------------------------------------------------------------------------
// Kernel 3: flash-attention pass. One wave per 16-row tile, streaming adj in
// 16x32 blocks; online softmax with rare-rescale fast path; P@H accumulated
// with v_wmma_f32_16x16x32_f16 (8 chunks of 16 output features).
// ---------------------------------------------------------------------------
__global__ __launch_bounds__(128) void k_flash(const _Float16* __restrict__ hT,
                                               const int* __restrict__ adj,
                                               const float* __restrict__ s1,
                                               const float* __restrict__ s2,
                                               float* __restrict__ out) {
    int wave = (blockIdx.x * blockDim.x + threadIdx.x) >> 5;
    int lane = threadIdx.x & 31;
    int n = lane & 15;
    int g = lane >> 4;

    int b  = wave / (T_DIM / 16);
    int i0 = (wave % (T_DIM / 16)) * 16;
    int nb = b * T_DIM;

    float rs1 = s1[nb + i0 + n];       // this lane's e-row (M = n)
    float m = -3.0e38f;                // running row max
    float l = 0.0f;                    // running row sum
    v8f acc[8];
#pragma unroll
    for (int c = 0; c < 8; ++c) acc[c] = v8f{};

    const int* arow = adj + ((size_t)(nb + i0 + n)) * T_DIM;
    const float* s2base = s2 + nb;

    // hoisted per-chunk hT base pointers (lane-constant across the j loop)
    const _Float16* hbase[8];
#pragma unroll
    for (int c = 0; c < 8; ++c)
        hbase[c] = hT + (size_t)(c * 16 + n) * NN + (size_t)nb;

    for (int j0 = 0; j0 < T_DIM; j0 += 32) {
        // adjacency: lane covers K in {8g..8g+7} U {16+8g..16+8g+7}
        const int* abase = arow + j0;
        int4 q0 = *(const int4*)(abase + 8 * g);
        int4 q1 = *(const int4*)(abase + 8 * g + 4);
        int4 q2 = *(const int4*)(abase + 16 + 8 * g);
        int4 q3 = *(const int4*)(abase + 20 + 8 * g);
        __builtin_prefetch(abase + 32, 0, 1);   // global_prefetch next block
        int am[16] = { q0.x, q0.y, q0.z, q0.w, q1.x, q1.y, q1.z, q1.w,
                       q2.x, q2.y, q2.z, q2.w, q3.x, q3.y, q3.z, q3.w };

        // s2 in slot order via direct vector loads (no bpermute chain):
        // slot e<8  -> K = e + 8g      -> s2base[j0 + 8g + e]
        // slot e>=8 -> K = e + 8 + 8g  -> s2base[j0 + 16 + 8g + (e-8)]
        const float* s2b = s2base + j0;
        float4 u0 = *(const float4*)(s2b + 8 * g);
        float4 u1 = *(const float4*)(s2b + 8 * g + 4);
        float4 u2 = *(const float4*)(s2b + 16 + 8 * g);
        float4 u3 = *(const float4*)(s2b + 20 + 8 * g);
        float s2s[16] = { u0.x, u0.y, u0.z, u0.w, u1.x, u1.y, u1.z, u1.w,
                          u2.x, u2.y, u2.z, u2.w, u3.x, u3.y, u3.z, u3.w };

        float ev[16];
        float rowmax = -3.0e38f;
#pragma unroll
        for (int e = 0; e < 16; ++e) {
            float t = rs1 + s2s[e];
            t = (t > 0.f) ? t : ALPHA * t;
            t = (am[e] == 0) ? MASKV : t;
            ev[e] = t;
            rowmax = fmaxf(rowmax, t);
        }
        rowmax = fmaxf(rowmax, __shfl_xor(rowmax, 16, 32));

        float mn = fmaxf(m, rowmax);
        float rs = 0.f;
        v16h P;
#pragma unroll
        for (int e = 0; e < 16; ++e) {
            float p = __expf(ev[e] - mn);
            rs += p;
            P[e] = (_Float16)p;
        }
        rs += __shfl_xor(rs, 16, 32);

        // fast path: no lane saw a new row max -> scale factor is exactly 1
        if (__ballot(mn > m)) {
            float scv = __expf(m - mn);     // first iter: exp(-inf) = 0
            l = l * scv + rs;
#pragma unroll
            for (int v = 0; v < 8; ++v) {
                float sv = __shfl(scv, v + 8 * g, 32);
#pragma unroll
                for (int c = 0; c < 8; ++c) acc[c][v] *= sv;
            }
        } else {
            l += rs;
        }
        m = mn;

        // P(16x32) @ H(32x16) per 16-feature chunk; B operand contiguous in hT
#pragma unroll
        for (int c = 0; c < 8; ++c) {
            const v8h* hp = (const v8h*)(hbase[c] + j0);
            v8h b0 = hp[g];
            v8h b1 = hp[2 + g];
            v16h Bv = __builtin_shufflevector(b0, b1, 0, 1, 2, 3, 4, 5, 6, 7,
                                              8, 9, 10, 11, 12, 13, 14, 15);
            acc[c] = __builtin_amdgcn_wmma_f32_16x16x32_f16(
                false, P, false, Bv, (short)0, acc[c], false, false);
        }
    }

    // epilogue: out[node][f] = acc / l  (l for row v+8g via bpermute)
#pragma unroll
    for (int v = 0; v < 8; ++v) {
        float lv  = __shfl(l, v + 8 * g, 32);
        float inv = 1.0f / lv;
        int row = i0 + v + 8 * g;
        float* orow = out + ((size_t)(nb + row)) * F_DIM;
#pragma unroll
        for (int c = 0; c < 8; ++c) orow[c * 16 + n] = acc[c][v] * inv;
    }
}

// ---------------------------------------------------------------------------
extern "C" void kernel_launch(void* const* d_in, const int* in_sizes, int n_in,
                              void* d_out, int out_size, void* d_ws, size_t ws_size,
                              hipStream_t stream) {
    const float* x   = (const float*)d_in[0];   // (B,T,128) f32
    const int*   adj = (const int*)d_in[1];     // (B,T,T) i32
    const float* W   = (const float*)d_in[2];   // (128,128) f32
    const float* a   = (const float*)d_in[3];   // (256,1) f32
    float* out = (float*)d_out;

    char* ws = (char*)d_ws;
    float*    Wt = (float*)ws;                                   // 64 KB
    _Float16* hT = (_Float16*)(ws + 65536);                      // 8 MB
    float*    s1 = (float*)(ws + 65536 + (size_t)F_DIM * NN * 2);
    float*    s2 = s1 + NN;

    k_twt   <<<F_DIM, F_DIM, 0, stream>>>(W, Wt);
    k_gemm_h<<<(NN / 16) / 4, 128, 0, stream>>>(x, Wt, hT);
    k_score <<<NN / 256, 256, 0, stream>>>(hT, a, s1, s2);
    k_flash <<<(NN / 16) / 4, 128, 0, stream>>>(hT, adj, s1, s2, out);
}
